// FastWeightCDE_84859963834872
// MI455X (gfx1250) — matrix operand order repaired and checked by reading, not compiled, not run on hardware
//
#include <hip/hip_runtime.h>
#include <hip/hip_bf16.h>

// ---------------- problem constants ----------------
#define B_    32
#define T_    32
#define CIN   64
#define HID   1024
#define NH    8
#define HD    128
#define COUT  128
#define FF_   4096
// unique f(t) stages: 32 knot evals + 31 (t0+h/3) + 31 (t0+2h/3)
#define NSTG  94
#define SB    (NSTG * B_)   // 3008 rows, multiple of 64
#define KC    128           // K chunk staged via TDM (multiple of 32; 128*2B = 256B rows)

typedef __hip_bfloat16 hbf;                                   // storage/convert type
typedef __bf16 bf16;                                          // WMMA element type
typedef __attribute__((ext_vector_type(16))) __bf16 v16bf;
typedef __attribute__((ext_vector_type(8)))  __bf16 v8bf;
typedef __attribute__((ext_vector_type(8)))  float  v8f;
typedef __attribute__((ext_vector_type(4)))  unsigned int u32x4;
typedef __attribute__((ext_vector_type(8)))  int          i32x8;
typedef __attribute__((ext_vector_type(4)))  int          i32x4;

// ---------------- WMMA helpers ----------------
__device__ __forceinline__ v8f wmma_bf16(v16bf a, v16bf b, v8f c) {
  return __builtin_amdgcn_wmma_f32_16x16x32_bf16(false, a, false, b, (short)0, c, false, false);
}
// A 16-bit fragment: lanes 0-15 K=0..15, lanes 16-31 K=16..31
__device__ __forceinline__ v16bf load_afrag(const bf16* row, int k0, int akb) {
  v16bf f;
  *((v8bf*)&f)     = *(const v8bf*)(row + k0 + akb);
  *((v8bf*)&f + 1) = *(const v8bf*)(row + k0 + 8 + akb);
  return f;
}

// ---------------- TDM: load a [64 x tile_k] bf16 tile (row stride K elems) into LDS ----------------
__device__ __forceinline__ void tdm_load_tile(const hbf* g, unsigned lds_off, int K, int tile_k) {
  unsigned long long ga = (unsigned long long)(uintptr_t)g;
  u32x4 g0 = { 1u,                                             // count=1, user mode
               lds_off,                                        // lds_addr (bytes)
               (unsigned)(ga & 0xffffffffu),                   // global_addr[31:0]
               (unsigned)((ga >> 32) & 0x01ffffffu) | (2u << 30) }; // addr[56:32] | type=2
  i32x8 g1;
  g1[0] = 0x00010000;                                         // data_size=1 (2 bytes), no flags
  g1[1] = (tile_k & 0xffff) << 16;                            // tensor_dim0[15:0]
  g1[2] = ((unsigned)tile_k >> 16) | (64u << 16);             // tensor_dim0[31:16] | tensor_dim1[15:0]=64
  g1[3] = (tile_k & 0xffff) << 16;                            // tensor_dim1[31:16]=0 | tile_dim0
  g1[4] = 64;                                                 // tile_dim1=64, tile_dim2=0
  g1[5] = K;                                                  // tensor_dim0_stride[31:0]
  g1[6] = 0;                                                  // stride0[47:32] | dim1_stride[15:0]
  g1[7] = 0;
  i32x4 z4 = {};
  i32x8 z8 = {};
  __builtin_amdgcn_tensor_load_to_lds(g0, g1, z4, z4, z8, 0); // 6-arg form (clang-23 lane)
}

// ---------------- bf16 WMMA GEMM (register blocked 32x64 per wave) ----------------
// C(MxN,f32) = A(MxK bf16 rowmajor) * BT(NxK bf16)^T + bias.  Block: 4 waves -> 64(M) x 128(N).
__global__ __launch_bounds__(128)
void gemm_bf16_wmma(const hbf* __restrict__ A, const hbf* __restrict__ BT,
                    const float* __restrict__ bias, float* __restrict__ C,
                    int M, int N, int K)
{
  const bf16* Ab = reinterpret_cast<const bf16*>(A);
  const bf16* Bb = reinterpret_cast<const bf16*>(BT);
  const int lane = threadIdx.x & 31;
  const int wave = threadIdx.x >> 5;
  const int m0 = blockIdx.y * 64 + (wave >> 1) * 32;
  const int n0 = blockIdx.x * 128 + (wave & 1) * 64;
  if (m0 >= M) return;                       // tail M (wave-uniform)
  const int mrow = lane & 15;
  const int akb  = (lane >> 4) * 16;
  const int bkb  = (lane >> 4) * 16;         // B 16-bit: lanes 0-15 K=0..15, lanes 16-31 K=16..31
  const bf16* ar0 = Ab + (size_t)(m0 + mrow) * K;
  const bf16* ar1 = ar0 + (size_t)16 * K;
  const bf16* br0 = Bb + (size_t)(n0 + mrow) * K;
  const bf16* br1 = br0 + (size_t)16 * K;
  const bf16* br2 = br0 + (size_t)32 * K;
  const bf16* br3 = br0 + (size_t)48 * K;
  v8f acc[2][4] = {};
  for (int k0 = 0; k0 < K; k0 += 32) {
    v16bf a0 = load_afrag(ar0, k0, akb);
    v16bf a1 = load_afrag(ar1, k0, akb);
    v16bf b0 = *(const v16bf*)(br0 + k0 + bkb);
    v16bf b1 = *(const v16bf*)(br1 + k0 + bkb);
    v16bf b2 = *(const v16bf*)(br2 + k0 + bkb);
    v16bf b3 = *(const v16bf*)(br3 + k0 + bkb);
    if (k0 + 32 < K) {
      __builtin_prefetch(ar0 + k0 + 32, 0, 3);
      __builtin_prefetch(br0 + k0 + 32, 0, 3);
    }
    acc[0][0] = wmma_bf16(a0, b0, acc[0][0]);
    acc[0][1] = wmma_bf16(a0, b1, acc[0][1]);
    acc[0][2] = wmma_bf16(a0, b2, acc[0][2]);
    acc[0][3] = wmma_bf16(a0, b3, acc[0][3]);
    acc[1][0] = wmma_bf16(a1, b0, acc[1][0]);
    acc[1][1] = wmma_bf16(a1, b1, acc[1][1]);
    acc[1][2] = wmma_bf16(a1, b2, acc[1][2]);
    acc[1][3] = wmma_bf16(a1, b3, acc[1][3]);
  }
  const int col = lane & 15;
  const int ro  = (lane >> 4) * 8;
  #pragma unroll
  for (int i = 0; i < 2; ++i) {
    #pragma unroll
    for (int j = 0; j < 4; ++j) {
      const int nn = n0 + j * 16 + col;
      const float bvv = bias ? bias[nn] : 0.0f;
      float* crow = C + (size_t)(m0 + i * 16 + ro) * N + nn;
      #pragma unroll
      for (int r = 0; r < 8; ++r)
        crow[(size_t)r * N] = acc[i][j][r] + bvv;
    }
  }
}

// ---------------- TDM-staged variant: A panel (64 x KC) double-buffered in LDS ----------------
// Requires: M % 64 == 0, N % 128 == 0, K % KC == 0.
__global__ __launch_bounds__(128)
void gemm_bf16_wmma_tdm(const hbf* __restrict__ A, const hbf* __restrict__ BT,
                        const float* __restrict__ bias, float* __restrict__ C,
                        int M, int N, int K)
{
  __shared__ hbf Abuf[2][64 * KC];           // 2 x 16KB
  const bf16* Bb = reinterpret_cast<const bf16*>(BT);
  const int lane = threadIdx.x & 31;
  const int wave = threadIdx.x >> 5;
  const int wid  = __builtin_amdgcn_readfirstlane(wave);   // scalar wave id -> s_cbranch gating
  const int m0 = blockIdx.y * 64;
  const int mloc = (wave >> 1) * 32;
  const int n0 = blockIdx.x * 128 + (wave & 1) * 64;
  const int mrow = lane & 15;
  const int akb  = (lane >> 4) * 16;
  const int bkb  = (lane >> 4) * 16;
  const hbf* Ag = A + (size_t)m0 * K;
  const bf16* br0 = Bb + (size_t)(n0 + mrow) * K;
  const bf16* br1 = br0 + (size_t)16 * K;
  const bf16* br2 = br0 + (size_t)32 * K;
  const bf16* br3 = br0 + (size_t)48 * K;
  const unsigned lds0 = (unsigned)(uintptr_t)(void*)&Abuf[0][0];
  const unsigned lds1 = (unsigned)(uintptr_t)(void*)&Abuf[1][0];
  const int nchunks = K / KC;

  if (wid == 0) tdm_load_tile(Ag, lds0, K, KC);            // chunk 0 -> buf 0
  v8f acc[2][4] = {};
  for (int c = 0; c < nchunks; ++c) {
    if (wid == 0) {
      if (c + 1 < nchunks) {
        tdm_load_tile(Ag + (size_t)(c + 1) * KC, ((c + 1) & 1) ? lds1 : lds0, K, KC);
        __builtin_amdgcn_s_wait_tensorcnt(1);              // oldest (chunk c) complete
      } else {
        __builtin_amdgcn_s_wait_tensorcnt(0);
      }
    }
    __syncthreads();
    const bf16* l0 = (const bf16*)&Abuf[c & 1][(mloc + mrow) * KC];
    const bf16* l1 = l0 + 16 * KC;
    const int kg = c * KC;
    #pragma unroll
    for (int kk = 0; kk < KC; kk += 32) {
      v16bf a0 = load_afrag(l0, kk, akb);
      v16bf a1 = load_afrag(l1, kk, akb);
      v16bf b0 = *(const v16bf*)(br0 + kg + kk + bkb);
      v16bf b1 = *(const v16bf*)(br1 + kg + kk + bkb);
      v16bf b2 = *(const v16bf*)(br2 + kg + kk + bkb);
      v16bf b3 = *(const v16bf*)(br3 + kg + kk + bkb);
      acc[0][0] = wmma_bf16(a0, b0, acc[0][0]);
      acc[0][1] = wmma_bf16(a0, b1, acc[0][1]);
      acc[0][2] = wmma_bf16(a0, b2, acc[0][2]);
      acc[0][3] = wmma_bf16(a0, b3, acc[0][3]);
      acc[1][0] = wmma_bf16(a1, b0, acc[1][0]);
      acc[1][1] = wmma_bf16(a1, b1, acc[1][1]);
      acc[1][2] = wmma_bf16(a1, b2, acc[1][2]);
      acc[1][3] = wmma_bf16(a1, b3, acc[1][3]);
    }
    __syncthreads();
  }
  const int col = lane & 15;
  const int ro  = (lane >> 4) * 8;
  #pragma unroll
  for (int i = 0; i < 2; ++i) {
    #pragma unroll
    for (int j = 0; j < 4; ++j) {
      const int nn = n0 + j * 16 + col;
      const float bvv = bias ? bias[nn] : 0.0f;
      float* crow = C + (size_t)(m0 + mloc + i * 16 + ro) * N + nn;
      #pragma unroll
      for (int r = 0; r < 8; ++r)
        crow[(size_t)r * N] = acc[i][j][r] + bvv;
    }
  }
}

// ---------------- stage time helper ----------------
__device__ __forceinline__ float stage_time(int s, const float* times) {
  if (s < 32) return times[s];
  if (s < 63) { int m = s - 31; return times[m-1] + (times[m] - times[m-1]) * (1.0f/3.0f); }
  int m = s - 62;              return times[m-1] + (times[m] - times[m-1]) * (2.0f/3.0f);
}

// ---------------- spline eval -> bf16 X, DX for all 94 stages ----------------
__global__ void spline_bf16(const float* __restrict__ times,
                            const float* __restrict__ ca, const float* __restrict__ cb,
                            const float* __restrict__ cc, const float* __restrict__ cd,
                            hbf* __restrict__ Xb, hbf* __restrict__ DXb)
{
  int i = blockIdx.x * 256 + threadIdx.x;
  if (i >= NSTG * B_ * CIN) return;
  int c  = i & (CIN - 1);
  int sb = i / CIN;
  int b  = sb & (B_ - 1);
  int s  = sb / B_;
  float t = stage_time(s, times);
  int idx = 0;                                   // searchsorted(right)-1, clipped to [0, T-2]
  for (int j = 1; j <= T_ - 2; ++j) if (times[j] <= t) idx = j;
  float fr = t - times[idx];
  size_t o = ((size_t)b * (T_ - 1) + idx) * CIN + c;
  float a = ca[o], bb = cb[o], c2 = cc[o], d3 = cd[o];
  float x  = a + (bb + (0.5f * c2 + d3 * fr * (1.0f/3.0f)) * fr) * fr;
  float dx = bb + (c2 + d3 * fr) * fr;
  Xb[i]  = __float2bfloat16(x);
  DXb[i] = __float2bfloat16(dx);
}

// ---------------- row LayerNorm (N=1024) -> bf16 ----------------
__global__ __launch_bounds__(256)
void layernorm_bf16(const float* __restrict__ X, const float* __restrict__ g,
                    const float* __restrict__ bta, hbf* __restrict__ Y, int N)
{
  const int row = blockIdx.x;
  const float* x = X + (size_t)row * N;
  __shared__ float red[16];
  __shared__ float mv[2];
  float s = 0.f, s2 = 0.f;
  for (int i = threadIdx.x; i < N; i += 256) { float v = x[i]; s += v; s2 = fmaf(v, v, s2); }
  #pragma unroll
  for (int off = 16; off; off >>= 1) { s += __shfl_down(s, off, 32); s2 += __shfl_down(s2, off, 32); }
  if ((threadIdx.x & 31) == 0) { red[threadIdx.x >> 5] = s; red[8 + (threadIdx.x >> 5)] = s2; }
  __syncthreads();
  if (threadIdx.x == 0) {
    float a = 0.f, b2 = 0.f;
    for (int i = 0; i < 8; ++i) { a += red[i]; b2 += red[8 + i]; }
    float mean = a / N;
    float var  = b2 / N - mean * mean;
    mv[0] = mean; mv[1] = rsqrtf(var + 1e-5f);
  }
  __syncthreads();
  const float mean = mv[0], inv = mv[1];
  hbf* y = Y + (size_t)row * N;
  for (int i = threadIdx.x; i < N; i += 256)
    y[i] = __float2bfloat16((x[i] - mean) * inv * g[i] + bta[i]);
}

// ---------------- softmax over contiguous groups of 128 (per b,h) ----------------
__global__ __launch_bounds__(128)
void softmax128(float* __restrict__ X)
{
  const size_t base = (size_t)blockIdx.x * 128;
  const int tid = threadIdx.x;
  __shared__ float red[4];
  float v = X[base + tid];
  float m = v;
  #pragma unroll
  for (int off = 16; off; off >>= 1) m = fmaxf(m, __shfl_down(m, off, 32));
  if ((tid & 31) == 0) red[tid >> 5] = m;
  __syncthreads();
  m = fmaxf(fmaxf(red[0], red[1]), fmaxf(red[2], red[3]));
  __syncthreads();
  float e = __expf(v - m);
  float ssum = e;
  #pragma unroll
  for (int off = 16; off; off >>= 1) ssum += __shfl_down(ssum, off, 32);
  if ((tid & 31) == 0) red[tid >> 5] = ssum;
  __syncthreads();
  float denom = red[0] + red[1] + red[2] + red[3];
  X[base + tid] = e / denom;
}

// ---------------- fast-weight collapse:
// r1[b,h,e] = sum_s w_s(b) * (v_s[b,h,:] . q[b,h,:]) * k_s[b,h,e] ----------------
__global__ __launch_bounds__(128)
void fastweight_reduce(const float* __restrict__ Kmat, const float* __restrict__ Vmat,
                       const float* __restrict__ Q, const float* __restrict__ times,
                       const int* __restrict__ fidx, float* __restrict__ R1)
{
  const int bh = blockIdx.x;                // 0..255
  const int b = bh >> 3, h = bh & 7;
  const int tid = threadIdx.x;              // 0..127 = d (dot) and e (output)
  const int F = fidx[b];
  const float q = Q[(size_t)b * HID + h * HD + tid];
  __shared__ float red[4];
  float racc = 0.0f;
  for (int s = 0; s < NSTG; ++s) {
    const size_t base = ((size_t)s * B_ + b) * HID + h * HD;
    float part = Vmat[base + tid] * q;
    #pragma unroll
    for (int off = 16; off; off >>= 1) part += __shfl_down(part, off, 32);
    if ((tid & 31) == 0) red[tid >> 5] = part;
    __syncthreads();
    const float alpha = red[0] + red[1] + red[2] + red[3];
    __syncthreads();
    float w;
    if (s == 0) {                                        // z0 (always) + k1 of step 1
      w = 1.0f + ((F >= 1) ? (times[1] - times[0]) * 0.125f : 0.0f);
    } else if (s <= 30) {                                // k4 of step s + k1 of step s+1
      w = 0.0f;
      if (F >= s)     w += (times[s]   - times[s-1]) * 0.125f;
      if (F >= s + 1) w += (times[s+1] - times[s]  ) * 0.125f;
    } else if (s == 31) {                                // k4 of final step
      w = (F >= 31) ? (times[31] - times[30]) * 0.125f : 0.0f;
    } else {                                             // k2 / k3 stages, weight 3h/8
      int m = (s <= 62) ? (s - 31) : (s - 62);
      w = (F >= m) ? 0.375f * (times[m] - times[m-1]) : 0.0f;
    }
    racc = fmaf(w * alpha, Kmat[base + tid], racc);
  }
  R1[(size_t)b * HID + h * HD + tid] = racc;
}

// ---------------- elementwise helpers ----------------
__global__ void pack_wT(const float* __restrict__ W, hbf* __restrict__ WT, int fin, int fout) {
  int i = blockIdx.x * 256 + threadIdx.x;
  if (i >= fin * fout) return;
  int n = i / fin, k = i - n * fin;
  WT[i] = __float2bfloat16(W[(size_t)k * fout + n]);       // WT[n][k] = W[k][n]
}
__global__ void f32_to_bf16_k(const float* __restrict__ X, hbf* __restrict__ Y, int n) {
  int i = blockIdx.x * 256 + threadIdx.x;
  if (i < n) Y[i] = __float2bfloat16(X[i]);
}
__global__ void tanh_k(float* __restrict__ X, int n) {
  int i = blockIdx.x * 256 + threadIdx.x;
  if (i < n) X[i] = tanhf(X[i]);
}
__global__ void relu_bf16_k(const float* __restrict__ X, hbf* __restrict__ Y, int n) {
  int i = blockIdx.x * 256 + threadIdx.x;
  if (i < n) Y[i] = __float2bfloat16(fmaxf(X[i], 0.0f));
}
__global__ void add_bf16_k(const float* __restrict__ Aa, const float* __restrict__ Bb2,
                           hbf* __restrict__ Y, int n) {
  int i = blockIdx.x * 256 + threadIdx.x;
  if (i < n) Y[i] = __float2bfloat16(Aa[i] + Bb2[i]);
}

// ---------------- host-side orchestration ----------------
extern "C" void kernel_launch(void* const* d_in, const int* in_sizes, int n_in,
                              void* d_out, int out_size, void* d_ws, size_t ws_size,
                              hipStream_t stream)
{
  (void)in_sizes; (void)n_in; (void)out_size; (void)ws_size;
  const float* times = (const float*)d_in[0];
  const float* ca    = (const float*)d_in[1];
  const float* cb    = (const float*)d_in[2];
  const float* cc    = (const float*)d_in[3];
  const float* cd    = (const float*)d_in[4];
  const float* Wi = (const float*)d_in[5];   const float* bi  = (const float*)d_in[6];
  const float* g_in=(const float*)d_in[7];   const float* b_in= (const float*)d_in[8];
  const float* Wd = (const float*)d_in[9];   const float* bd  = (const float*)d_in[10];
  const float* Wk = (const float*)d_in[11];  const float* bk  = (const float*)d_in[12];
  const float* Wv = (const float*)d_in[13];  const float* bv  = (const float*)d_in[14];
  const float* Wq = (const float*)d_in[15];  const float* bq  = (const float*)d_in[16];
  const float* Wo = (const float*)d_in[17];  const float* bo  = (const float*)d_in[18];
  const float* g_ff=(const float*)d_in[19];  const float* b_ff=(const float*)d_in[20];
  const float* W1 = (const float*)d_in[21];  const float* b1  = (const float*)d_in[22];
  const float* W2 = (const float*)d_in[23];  const float* b2  = (const float*)d_in[24];
  const float* Wl = (const float*)d_in[25];  const float* bl  = (const float*)d_in[26];
  const int*   fidx = (const int*)d_in[27];

  char* p = (char*)d_ws;
  auto alloc = [&](size_t bytes) -> void* {
    void* r = (void*)p; p += (bytes + 255) & ~(size_t)255; return r;
  };
  hbf* WiT = (hbf*)alloc(sizeof(hbf) * (size_t)HID * CIN);
  hbf* WdT = (hbf*)alloc(sizeof(hbf) * (size_t)HID * CIN);
  hbf* WkT = (hbf*)alloc(sizeof(hbf) * (size_t)HID * HID);
  hbf* WvT = (hbf*)alloc(sizeof(hbf) * (size_t)HID * HID);
  hbf* WqT = (hbf*)alloc(sizeof(hbf) * (size_t)HID * HID);
  hbf* WoT = (hbf*)alloc(sizeof(hbf) * (size_t)HID * HID);
  hbf* W1T = (hbf*)alloc(sizeof(hbf) * (size_t)FF_ * HID);
  hbf* W2T = (hbf*)alloc(sizeof(hbf) * (size_t)HID * FF_);
  hbf* WlT = (hbf*)alloc(sizeof(hbf) * (size_t)COUT * HID);
  hbf* Xb    = (hbf*)alloc(sizeof(hbf) * (size_t)SB * CIN);
  hbf* DXb   = (hbf*)alloc(sizeof(hbf) * (size_t)SB * CIN);
  float* Hpre  = (float*)alloc(sizeof(float) * (size_t)SB * HID);
  float* HDpre = (float*)alloc(sizeof(float) * (size_t)SB * HID);
  hbf*   Hb    = (hbf*)  alloc(sizeof(hbf)   * (size_t)SB * HID);
  hbf*   HDb   = (hbf*)  alloc(sizeof(hbf)   * (size_t)SB * HID);
  float* Kmat  = (float*)alloc(sizeof(float) * (size_t)SB * HID);
  float* Vmat  = (float*)alloc(sizeof(float) * (size_t)SB * HID);
  float* Qm    = (float*)alloc(sizeof(float) * (size_t)B_ * HID);
  float* R1    = (float*)alloc(sizeof(float) * (size_t)B_ * HID);
  hbf*   R1b   = (hbf*)  alloc(sizeof(hbf)   * (size_t)B_ * HID);
  float* Ro    = (float*)alloc(sizeof(float) * (size_t)B_ * HID);
  hbf*   Tb    = (hbf*)  alloc(sizeof(hbf)   * (size_t)B_ * HID);
  float* U     = (float*)alloc(sizeof(float) * (size_t)B_ * FF_);
  hbf*   Ub    = (hbf*)  alloc(sizeof(hbf)   * (size_t)B_ * FF_);
  float* R2    = (float*)alloc(sizeof(float) * (size_t)B_ * HID);
  hbf*   Rfb   = (hbf*)  alloc(sizeof(hbf)   * (size_t)B_ * HID);

  auto packT = [&](const float* W, hbf* WT, int fin, int fout) {
    int n = fin * fout;
    pack_wT<<<(n + 255) / 256, 256, 0, stream>>>(W, WT, fin, fout);
  };
  auto gemm = [&](const hbf* A, const hbf* BT, const float* bias, float* C,
                  int M, int N, int K) {
    gemm_bf16_wmma<<<dim3(N / 128, (M + 63) / 64), dim3(128), 0, stream>>>(A, BT, bias, C, M, N, K);
  };
  auto gemm_tdm = [&](const hbf* A, const hbf* BT, const float* bias, float* C,
                      int M, int N, int K) {   // M%64==0, N%128==0, K%KC==0
    gemm_bf16_wmma_tdm<<<dim3(N / 128, M / 64), dim3(128), 0, stream>>>(A, BT, bias, C, M, N, K);
  };

  // 1. weight packing (f32 -> bf16, transposed)
  packT(Wi, WiT, CIN, HID);  packT(Wd, WdT, CIN, HID);
  packT(Wk, WkT, HID, HID);  packT(Wv, WvT, HID, HID);
  packT(Wq, WqT, HID, HID);  packT(Wo, WoT, HID, HID);
  packT(W1, W1T, HID, FF_);  packT(W2, W2T, FF_, HID);
  packT(Wl, WlT, HID, COUT);

  // 2. spline eval for all 94 stages
  {
    int n = NSTG * B_ * CIN;
    spline_bf16<<<(n + 255) / 256, 256, 0, stream>>>(times, ca, cb, cc, cd, Xb, DXb);
  }

  // 3. stage GEMMs (batched across all stages — serial scan collapsed)
  gemm(Xb,  WiT, bi, Hpre,  SB, HID, CIN);             // x @ Wi + bi   (K=64)
  gemm(DXb, WdT, bd, HDpre, SB, HID, CIN);             // dx @ Wd + bd  (K=64)
  layernorm_bf16<<<SB, 256, 0, stream>>>(Hpre, g_in, b_in, Hb, HID);
  f32_to_bf16_k<<<(SB * HID + 255) / 256, 256, 0, stream>>>(HDpre, HDb, SB * HID);
  gemm_tdm(Hb,  WkT, bk, Kmat, SB, HID, HID);          // h @ Wk + bk  (TDM-staged A)
  softmax128<<<SB * NH, 128, 0, stream>>>(Kmat);       // softmax per (s,b,h)
  gemm_tdm(HDb, WvT, bv, Vmat, SB, HID, HID);          // hd @ Wv + bv (TDM-staged A)
  tanh_k<<<(SB * HID + 255) / 256, 256, 0, stream>>>(Vmat, SB * HID);

  // 4. query path: h at t_last is knot stage s=31 (already LayerNormed)
  gemm(Hb + (size_t)31 * B_ * HID, WqT, bq, Qm, B_, HID, HID);
  softmax128<<<B_ * NH, 128, 0, stream>>>(Qm);

  // 5. fast-weight readout (collapses z scan + einsum into rank-1 sum)
  fastweight_reduce<<<B_ * NH, 128, 0, stream>>>(Kmat, Vmat, Qm, times, fidx, R1);

  // 6. output head: Wo, FFN residual, Wl
  f32_to_bf16_k<<<(B_ * HID + 255) / 256, 256, 0, stream>>>(R1, R1b, B_ * HID);
  gemm(R1b, WoT, bo, Ro, B_, HID, HID);                // r @ Wo + bo
  layernorm_bf16<<<B_, 256, 0, stream>>>(Ro, g_ff, b_ff, Tb, HID);
  gemm(Tb, W1T, b1, U, B_, FF_, HID);                  // LN(r) @ W1 + b1
  relu_bf16_k<<<(B_ * FF_ + 255) / 256, 256, 0, stream>>>(U, Ub, B_ * FF_);
  gemm(Ub, W2T, b2, R2, B_, HID, FF_);                 // relu @ W2 + b2
  add_bf16_k<<<(B_ * HID + 255) / 256, 256, 0, stream>>>(Ro, R2, Rfb, B_ * HID);
  gemm(Rfb, WlT, bl, (float*)d_out, B_, COUT, HID);    // final logits (32 x 128 f32)
}
